// MoETransformer_59562606460993
// MI455X (gfx1250) — matrix operand order repaired and checked by reading, not compiled, not run on hardware
//
#include <hip/hip_runtime.h>
#include <cstdint>

#define HD   768
#define NHD  12
#define LL   4
#define EE   8
#define GG   4
#define DD   64
#define FF   3072
#define SS   1024
#define VV   50257
#define GDD  (GG*DD)

typedef __attribute__((ext_vector_type(16))) __bf16 bf16x16;
typedef __attribute__((ext_vector_type(8)))  float  f32x8;

// 16-byte load unit that only promises 4-byte alignment (vocab GEMM has odd ldb)
struct __attribute__((packed, aligned(4))) F4 { float x, y, z, w; };

// single-instruction f32x2 -> packed bf16 (dst.lo = cvt(lo), dst.hi = cvt(hi))
static __device__ __forceinline__ unsigned pack2(float lo, float hi) {
  unsigned r;
  asm("v_cvt_pk_bf16_f32 %0, %1, %2" : "=v"(r) : "v"(lo), "v"(hi));
  return r;
}

union Frag { unsigned u[8]; bf16x16 v; };

// --- CDNA5 async global<->LDS helpers (ASYNCcnt-tracked) --------------------
static __device__ __forceinline__ void async_load_b128_to_lds(unsigned lds_off,
                                                              const void* gaddr) {
  asm volatile("global_load_async_to_lds_b128 %0, %1, off"
               :: "v"(lds_off), "v"(gaddr) : "memory");
}
static __device__ __forceinline__ void async_store_b128_from_lds(void* gaddr,
                                                                 unsigned lds_off) {
  asm volatile("global_store_async_from_lds_b128 %0, %1, off"
               :: "v"(gaddr), "v"(lds_off) : "memory");
}
static __device__ __forceinline__ void wait_async0() {
  asm volatile("s_wait_asynccnt 0x0" ::: "memory");
}
static __device__ __forceinline__ void wait_ds0() {
  asm volatile("s_wait_dscnt 0x0" ::: "memory");
}

// ---------------------------------------------------------------------------
// Generic WMMA bf16 GEMM: C = epilogue(alpha * A[M,K] x B + bias)
// Block tile 64x128, BK=32, 256 threads (8 waves), wave tile 32x32 (2x2 WMMA).
// TRANSB     : B given as [N,K] row-major (use B^T)
// GELU_ACT   : exact-erf gelu on the result
// GATHER_A   : A row m is A[rowIdx[bm+m]]  (MoE fc1 token gather)
// SCATTER_ACC: C[rowIdx[row]] += gates[row] * val  (MoE fc2 scatter-accum)
// GATHER_A||SCATTER_ACC: early-exit blocks with bm >= ceil(*cntPtr/64)*64
// Assumes M % 64 == 0 and K % 32 == 0 (true for every call here).
// ---------------------------------------------------------------------------
template<bool TRANSB, bool GELU_ACT, bool GATHER_A, bool SCATTER_ACC>
__global__ __launch_bounds__(256)
void gemm_bf16_wmma(const float* __restrict__ A, const float* __restrict__ B,
                    const float* __restrict__ bias,
                    const int* __restrict__ rowIdx, const float* __restrict__ gates,
                    const int* __restrict__ cntPtr,
                    float* __restrict__ C,
                    int M, int N, int K, int lda, int ldb, int ldc,
                    long bsA, long bsB, long bsC, int bDiv, float alpha)
{
  __shared__ unsigned As[64][17];   // [m][k/2] packed bf16 pair
  __shared__ unsigned Bs[128][17];  // [n][k/2] packed bf16 pair

  const int tid = threadIdx.x;
  const int bm  = blockIdx.y * 64;
  const int bn  = blockIdx.x * 128;
  const int z   = blockIdx.z;
  A += (long)z * bsA;
  B += (long)(z / bDiv) * bsB;
  C += (long)z * bsC;

  if (GATHER_A || SCATTER_ACC) {
    int cnt = *cntPtr;                     // scalar load, uniform
    if (bm >= ((cnt + 63) & ~63)) return;  // beyond padded token count
  }

  const int lane  = tid & 31;
  const int wave  = tid >> 5;
  const int wm    = wave >> 2;   // 0..1
  const int wn    = wave & 3;    // 0..3
  const int lhalf = lane >> 4;   // 0..1
  const int l16   = lane & 15;

  // fixed per-thread staging coordinates
  const int am  = tid >> 2;            // A: row 0..63
  const int ap0 = (tid & 3) * 4;       // A: first packed index {0,4,8,12}
  long arow0 = GATHER_A ? (long)rowIdx[bm + am] : (long)(bm + am);

  f32x8 acc[2][2] = {};

  for (int k0 = 0; k0 < K; k0 += 32) {
    // ---- stage A tile (64x32): 2x b128 loads, 4 cvt_pk, 4 contiguous stores
    {
      const F4* ap = (const F4*)(A + arow0 * lda + k0 + 2 * ap0);
      F4 f0 = ap[0], f1 = ap[1];
      As[am][ap0 + 0] = pack2(f0.x, f0.y);
      As[am][ap0 + 1] = pack2(f0.z, f0.w);
      As[am][ap0 + 2] = pack2(f1.x, f1.y);
      As[am][ap0 + 3] = pack2(f1.z, f1.w);
    }
    // ---- stage B tile (32x128)
    if (TRANSB) {
      int n  = tid >> 1;                 // 0..127
      int p0 = (tid & 1) * 8;            // 0 or 8
      int gn = bn + n;
      if (gn < N) {
        const F4* bp = (const F4*)(B + (long)gn * ldb + k0 + 2 * p0);
        F4 f0 = bp[0], f1 = bp[1], f2 = bp[2], f3 = bp[3];
        Bs[n][p0 + 0] = pack2(f0.x, f0.y);
        Bs[n][p0 + 1] = pack2(f0.z, f0.w);
        Bs[n][p0 + 2] = pack2(f1.x, f1.y);
        Bs[n][p0 + 3] = pack2(f1.z, f1.w);
        Bs[n][p0 + 4] = pack2(f2.x, f2.y);
        Bs[n][p0 + 5] = pack2(f2.z, f2.w);
        Bs[n][p0 + 6] = pack2(f3.x, f3.y);
        Bs[n][p0 + 7] = pack2(f3.z, f3.w);
      } else {
#pragma unroll
        for (int j = 0; j < 8; ++j) Bs[n][p0 + j] = 0u;
      }
    } else {
#pragma unroll
      for (int i = 0; i < 2; ++i) {
        int g   = tid + i * 256;         // 0..511 pack-groups of 4 cols
        int p   = g >> 5;                // 0..15
        int n0  = (g & 31) * 4;
        int gn0 = bn + n0;
        const float* brow = B + (long)(k0 + 2 * p) * ldb;
        if (gn0 + 3 < N) {
          F4 f0 = *(const F4*)(brow + gn0);
          F4 f1 = *(const F4*)(brow + ldb + gn0);
          Bs[n0 + 0][p] = pack2(f0.x, f1.x);
          Bs[n0 + 1][p] = pack2(f0.y, f1.y);
          Bs[n0 + 2][p] = pack2(f0.z, f1.z);
          Bs[n0 + 3][p] = pack2(f0.w, f1.w);
        } else {
#pragma unroll
          for (int j = 0; j < 4; ++j) {
            int gn = gn0 + j;
            Bs[n0 + j][p] = (gn < N) ? pack2(brow[gn], brow[ldb + gn]) : 0u;
          }
        }
      }
    }
    __syncthreads();

    // light prefetch of next K tile (global_prefetch_b8)
    if (k0 + 32 < K) {
      __builtin_prefetch(A + arow0 * lda + k0 + 32 + 2 * ap0, 0, 1);
      if (!TRANSB)
        __builtin_prefetch(B + (long)(k0 + 32 + (tid >> 6)) * ldb + bn + (tid & 63) * 2, 0, 1);
    }

    // ---- build fragments per documented gfx1250 layouts
    Frag a[2], b[2];
#pragma unroll
    for (int i = 0; i < 2; ++i) {
      int m = wm * 32 + i * 16 + l16;
#pragma unroll
      for (int j = 0; j < 4; ++j) {
        a[i].u[j]     = As[m][lhalf * 4 + j];        // K 0..7 / 8..15
        a[i].u[j + 4] = As[m][8 + lhalf * 4 + j];    // K 16..23 / 24..31
      }
    }
#pragma unroll
    for (int i = 0; i < 2; ++i) {
      int n = wn * 32 + i * 16 + l16;
#pragma unroll
      for (int j = 0; j < 4; ++j) {
        b[i].u[j]     = Bs[n][lhalf * 4 + j];        // K 0..7 / 8..15
        b[i].u[j + 4] = Bs[n][8 + lhalf * 4 + j];    // K 16..23 / 24..31
      }
    }
#pragma unroll
    for (int i = 0; i < 2; ++i)
#pragma unroll
      for (int j = 0; j < 2; ++j)
        acc[i][j] = __builtin_amdgcn_wmma_f32_16x16x32_bf16(
            false, a[i].v, false, b[j].v, (short)0, acc[i][j], false, false);

    __syncthreads();
  }

  // ---- epilogue
#pragma unroll
  for (int i = 0; i < 2; ++i) {
    int rowb = bm + wm * 32 + i * 16 + lhalf * 8;
#pragma unroll
    for (int j = 0; j < 2; ++j) {
      int col = bn + wn * 32 + j * 16 + l16;
      if (col < N) {
#pragma unroll
        for (int r = 0; r < 8; ++r) {
          int row = rowb + r;
          float v = acc[i][j][r] * alpha;
          if (bias) v += bias[col];
          if (GELU_ACT) v = 0.5f * v * (1.0f + erff(v * 0.70710678118654752f));
          if (SCATTER_ACC) {
            long orow = (long)rowIdx[row];
            C[orow * ldc + col] += gates[row] * v;   // unique rows per expert
          } else {
            C[(long)row * ldc + col] = v;
          }
        }
      }
    }
  }
}

// ---------------------------------------------------------------------------
__global__ __launch_bounds__(256)
void embed_kernel(const int* __restrict__ ids, const float* __restrict__ emb,
                  const float* __restrict__ pos, float* __restrict__ h)
{
  int s = blockIdx.x;
  int id = ids[s];
  for (int c = threadIdx.x; c < HD; c += 256)
    h[(long)s * HD + c] = emb[(long)id * HD + c] + pos[(long)s * HD + c];
}

// Softmax over rows of length 1024: stage the row to LDS with async B128
// copies (read once instead of 3x), compute in LDS, async-store back.
__global__ __launch_bounds__(256)
void softmax_rows_async(float* __restrict__ data)
{
  __shared__ float buf[1024];
  __shared__ float red[8];
  long row = blockIdx.x;
  float* p = data + row * 1024;
  const int tid = threadIdx.x;

  unsigned loff = (unsigned)(uintptr_t)buf + (unsigned)tid * 16u;
  async_load_b128_to_lds(loff, p + tid * 4);
  wait_async0();
  __syncthreads();

  // row max
  float m = fmaxf(fmaxf(buf[tid * 4 + 0], buf[tid * 4 + 1]),
                  fmaxf(buf[tid * 4 + 2], buf[tid * 4 + 3]));
  for (int o = 16; o; o >>= 1) m = fmaxf(m, __shfl_xor(m, o, 32));
  if ((tid & 31) == 0) red[tid >> 5] = m;
  __syncthreads();
  float bm = red[0];
#pragma unroll
  for (int i = 1; i < 8; ++i) bm = fmaxf(bm, red[i]);
  __syncthreads();

  // exp + sum
  float s = 0.0f;
#pragma unroll
  for (int j = 0; j < 4; ++j) {
    float e = expf(buf[tid * 4 + j] - bm);
    buf[tid * 4 + j] = e;
    s += e;
  }
  for (int o = 16; o; o >>= 1) s += __shfl_xor(s, o, 32);
  if ((tid & 31) == 0) red[tid >> 5] = s;
  __syncthreads();
  float tot = 0.0f;
#pragma unroll
  for (int i = 0; i < 8; ++i) tot += red[i];
  float inv = 1.0f / tot;

  // scale in LDS, then async store this lane's 16B back to global
#pragma unroll
  for (int j = 0; j < 4; ++j) buf[tid * 4 + j] *= inv;
  wait_ds0();                       // ensure LDS writes landed before async read
  async_store_b128_from_lds(p + tid * 4, loff);
  wait_async0();
}

// out = LayerNorm(x + y) * g + b   (one row per block; safe for out==x)
__global__ __launch_bounds__(256)
void add_layernorm(const float* __restrict__ x, const float* __restrict__ y,
                   const float* __restrict__ g, const float* __restrict__ b,
                   float* __restrict__ out)
{
  long row = blockIdx.x;
  int tid = threadIdx.x;
  __shared__ float red[8], red2[8];
  float v[3]; int idx[3]; int nv = 0;
  float s = 0.0f, sq = 0.0f;
  for (int c = tid; c < HD; c += 256) {
    float t = x[row * HD + c] + y[row * HD + c];
    v[nv] = t; idx[nv] = c; ++nv;
    s += t; sq += t * t;
  }
  for (int o = 16; o; o >>= 1) { s += __shfl_xor(s, o, 32); sq += __shfl_xor(sq, o, 32); }
  if ((tid & 31) == 0) { red[tid >> 5] = s; red2[tid >> 5] = sq; }
  __syncthreads();
  float S = 0.0f, SQ = 0.0f;
#pragma unroll
  for (int i = 0; i < 8; ++i) { S += red[i]; SQ += red2[i]; }
  float mu  = S * (1.0f / HD);
  float var = SQ * (1.0f / HD) - mu * mu;
  float rs  = rsqrtf(var + 1e-5f);
  for (int i = 0; i < nv; ++i)
    out[row * HD + idx[i]] = (v[i] - mu) * rs * g[idx[i]] + b[idx[i]];
}

__global__ __launch_bounds__(256)
void router_logits(const float* __restrict__ h, const float* __restrict__ rw,
                   const float* __restrict__ rb, float* __restrict__ out)
{
  int idx = blockIdx.x * 256 + threadIdx.x;      // token*8 + e
  if (idx >= SS * EE) return;
  int t = idx >> 3, e = idx & 7;
  float s = rb[e];
  const float* hp = h + (long)t * HD;
  for (int k = 0; k < HD; ++k) s += hp[k] * rw[(long)k * EE + e];
  out[idx] = s;
}

// Top-2 routing: append token to each chosen expert's compact list with its
// gate. (expIdx/expGate pre-zeroed: pad rows read token 0 with gate 0.)
__global__ __launch_bounds__(256)
void top2_scatter(const float* __restrict__ rl, int* __restrict__ expIdx,
                  float* __restrict__ expGate, int* __restrict__ cnt)
{
  int t = blockIdx.x * 256 + threadIdx.x;
  if (t >= SS) return;
  float lg[EE], p[EE];
  float mx = -3.4e38f;
#pragma unroll
  for (int e = 0; e < EE; ++e) { lg[e] = rl[t * EE + e]; mx = fmaxf(mx, lg[e]); }
  float s = 0.0f;
#pragma unroll
  for (int e = 0; e < EE; ++e) { p[e] = expf(lg[e] - mx); s += p[e]; }
  float inv = 1.0f / s;
#pragma unroll
  for (int e = 0; e < EE; ++e) p[e] *= inv;
  int i1 = 0;
#pragma unroll
  for (int e = 1; e < EE; ++e) if (p[e] > p[i1]) i1 = e;
  int i2 = (i1 == 0) ? 1 : 0;
#pragma unroll
  for (int e = 0; e < EE; ++e) if (e != i1 && p[e] > p[i2]) i2 = e;
  // reference quirk: gates = softmax over the two top *probabilities*
  float g1 = expf(p[i1]), g2 = expf(p[i2]);
  float gs = 1.0f / (g1 + g2);
  int s1 = atomicAdd(&cnt[i1], 1);
  expIdx [i1 * SS + s1] = t;
  expGate[i1 * SS + s1] = g1 * gs;
  int s2 = atomicAdd(&cnt[i2], 1);
  expIdx [i2 * SS + s2] = t;
  expGate[i2 * SS + s2] = g2 * gs;
}

__global__ void moe_loss_kernel(const int* __restrict__ counts, float* __restrict__ out)
{
  if (threadIdx.x == 0 && blockIdx.x == 0) {
    float mean = 0.0f;
#pragma unroll
    for (int e = 0; e < EE; ++e) mean += (float)counts[e];
    mean *= (1.0f / EE);
    float var = 0.0f;
#pragma unroll
    for (int e = 0; e < EE; ++e) { float d = (float)counts[e] - mean; var += d * d; }
    var *= (1.0f / (EE - 1));
    out[0] = var / (mean * mean);
  }
}

// ---------------------------------------------------------------------------
extern "C" void kernel_launch(void* const* d_in, const int* in_sizes, int n_in,
                              void* d_out, int out_size, void* d_ws, size_t ws_size,
                              hipStream_t stream)
{
  (void)in_sizes; (void)n_in; (void)out_size; (void)ws_size;
  const int*   ids  = (const int*)  d_in[0];
  // d_in[1] attention_mask: all ones -> additive mask == 0 everywhere
  const float* emb  = (const float*)d_in[2];
  const float* pos  = (const float*)d_in[3];
  const float* qw   = (const float*)d_in[4];
  const float* qb   = (const float*)d_in[5];
  const float* kw   = (const float*)d_in[6];
  const float* kb   = (const float*)d_in[7];
  const float* vw   = (const float*)d_in[8];
  const float* vb   = (const float*)d_in[9];
  const float* rw   = (const float*)d_in[10];
  const float* rb   = (const float*)d_in[11];
  const float* f1w  = (const float*)d_in[12];
  const float* f1b  = (const float*)d_in[13];
  const float* f2w  = (const float*)d_in[14];
  const float* f2b  = (const float*)d_in[15];
  const float* lng  = (const float*)d_in[16];
  const float* lnb  = (const float*)d_in[17];
  const float* ow   = (const float*)d_in[18];
  const float* ob   = (const float*)d_in[19];

  float* logits = (float*)d_out;                 // [S,V]
  float* losses = logits + (long)SS * VV;        // [L]

  // workspace layout
  float* ws = (float*)d_ws;
  float* h       = ws;                          // 786432
  float* q       = h       + (long)SS * HD;     // 786432
  float* kbuf    = q       + (long)SS * HD;     // 262144
  float* vbuf    = kbuf    + (long)SS * GDD;    // 262144
  float* ao      = vbuf    + (long)SS * GDD;    // 786432
  float* moe     = ao      + (long)SS * HD;     // 786432
  float* h1      = moe     + (long)SS * HD;     // 3145728 (compact per-expert)
  float* rlog    = h1      + (long)SS * FF;     // 8192
  float* expGate = rlog    + (long)SS * EE;     // 8192
  int*   expIdx  = (int*)  (expGate + (long)SS * EE);  // 8192 ints
  int*   cnt     = expIdx  + (long)SS * EE;     // 8 ints
  float* scores  = (float*)(cnt + 16);          // NH*S*S = 12582912

  dim3 blk(256);
  const float attn_scale = 0.125f;  // 1/sqrt(64)

  embed_kernel<<<SS, blk, 0, stream>>>(ids, emb, pos, h);

  for (int l = 0; l < LL; ++l) {
    // ---- QKV projections
    gemm_bf16_wmma<false,false,false,false><<<dim3(HD/128, SS/64, 1), blk, 0, stream>>>(
        h, qw + (long)l * HD * HD, qb + (long)l * HD, nullptr, nullptr, nullptr, q,
        SS, HD, HD, HD, HD, HD, 0, 0, 0, 1, 1.0f);
    gemm_bf16_wmma<false,false,false,false><<<dim3(GDD/128, SS/64, 1), blk, 0, stream>>>(
        h, kw + (long)l * HD * GDD, kb + (long)l * GDD, nullptr, nullptr, nullptr, kbuf,
        SS, GDD, HD, HD, GDD, GDD, 0, 0, 0, 1, 1.0f);
    gemm_bf16_wmma<false,false,false,false><<<dim3(GDD/128, SS/64, 1), blk, 0, stream>>>(
        h, vw + (long)l * HD * GDD, vb + (long)l * GDD, nullptr, nullptr, nullptr, vbuf,
        SS, GDD, HD, HD, GDD, GDD, 0, 0, 0, 1, 1.0f);

    // ---- scores[h] = (q_h @ k_{h/3}^T) * scale, batched over 12 heads
    gemm_bf16_wmma<true,false,false,false><<<dim3(SS/128, SS/64, NHD), blk, 0, stream>>>(
        q, kbuf, nullptr, nullptr, nullptr, nullptr, scores,
        SS, SS, DD, HD, GDD, SS,
        /*bsA=*/DD, /*bsB=*/DD, /*bsC=*/(long)SS * SS, /*bDiv=*/NHD/GG, attn_scale);

    softmax_rows_async<<<NHD * SS, blk, 0, stream>>>(scores);

    // ---- ao[:, h*64:(h+1)*64] = attn_h @ v_{h/3}
    gemm_bf16_wmma<false,false,false,false><<<dim3(1, SS/64, NHD), blk, 0, stream>>>(
        scores, vbuf, nullptr, nullptr, nullptr, nullptr, ao,
        SS, DD, SS, SS, GDD, HD,
        /*bsA=*/(long)SS * SS, /*bsB=*/DD, /*bsC=*/DD, /*bDiv=*/NHD/GG, 1.0f);

    add_layernorm<<<SS, blk, 0, stream>>>(h, ao, lng, lnb, h);

    // ---- router + top-2 token scatter + load-balance loss
    router_logits<<<(SS * EE + 255) / 256, blk, 0, stream>>>(
        h, rw + (long)l * HD * EE, rb + (long)l * EE, rlog);
    hipMemsetAsync(cnt, 0, EE * sizeof(int), stream);
    hipMemsetAsync(expIdx, 0, (long)SS * EE * sizeof(int), stream);
    hipMemsetAsync(expGate, 0, (long)SS * EE * sizeof(float), stream);
    top2_scatter<<<(SS + 255) / 256, blk, 0, stream>>>(rlog, expIdx, expGate, cnt);
    moe_loss_kernel<<<1, 1, 0, stream>>>(cnt, losses + l);

    // ---- experts: gathered fc1 (gelu) + scatter-accumulate fc2
    hipMemsetAsync(moe, 0, (long)SS * HD * sizeof(float), stream);
    for (int e = 0; e < EE; ++e) {
      const float* w1 = f1w + ((long)l * EE + e) * HD * FF;
      const float* b1 = f1b + ((long)l * EE + e) * FF;
      const float* w2 = f2w + ((long)l * EE + e) * FF * HD;
      const float* b2 = f2b + ((long)l * EE + e) * HD;
      gemm_bf16_wmma<false,true,true,false><<<dim3(FF/128, SS/64, 1), blk, 0, stream>>>(
          h, w1, b1, expIdx + (long)e * SS, nullptr, cnt + e, h1,
          SS, FF, HD, HD, FF, FF, 0, 0, 0, 1, 1.0f);
      gemm_bf16_wmma<false,false,false,true><<<dim3(HD/128, SS/64, 1), blk, 0, stream>>>(
          h1, w2, b2, expIdx + (long)e * SS, expGate + (long)e * SS, cnt + e, moe,
          SS, HD, FF, FF, HD, HD, 0, 0, 0, 1, 1.0f);
    }

    add_layernorm<<<SS, blk, 0, stream>>>(h, moe, lng, lnb, h);
  }

  // ---- final vocab projection into d_out
  gemm_bf16_wmma<false,false,false,false><<<dim3((VV + 127) / 128, SS/64, 1), blk, 0, stream>>>(
      h, ow, ob, nullptr, nullptr, nullptr, logits,
      SS, VV, HD, HD, VV, VV, 0, 0, 0, 1, 1.0f);
}
